// NewAttention_33182917329411
// MI455X (gfx1250) — matrix operand order repaired and checked by reading, not compiled, not run on hardware
//
#include <hip/hip_runtime.h>

// ---------------------------------------------------------------------------
// NewAttention on gfx1250 (MI455X):
//   v        = x @ W_in^T        f16 WMMA GEMM, LDS double-buffered, fp32 acc
//   attended = gaussian conv(v)  sigma=1 -> +/-11 taps, constant weights
//   out      = attended @ W_out^T                      (fp32 out)
// ---------------------------------------------------------------------------

typedef _Float16 f16;
typedef _Float16 v16h __attribute__((ext_vector_type(16)));
typedef _Float16 v8h  __attribute__((ext_vector_type(8)));
typedef float    v8f  __attribute__((ext_vector_type(8)));
typedef int      v4i  __attribute__((ext_vector_type(4)));

#define BATCH   8
#define SEQLEN  2048
#define EMBED   1024
#define RADIUS  11          // exp(-11^2/2) ~ 5e-27: below fp32 significance
#define BK      32          // K step (one 16x16x32 WMMA depth)
#define LDSROW  40          // halves per LDS row: 32 data + 8 pad (80B = 20 banks)

#if __has_builtin(__builtin_amdgcn_global_load_async_to_lds_b128)
#define ASYNC_LDS 1
#else
#define ASYNC_LDS 0
#endif

#if ASYNC_LDS
__device__ __forceinline__ void async_cp16(const f16* g, f16* l) {
    __builtin_amdgcn_global_load_async_to_lds_b128(
        (__attribute__((address_space(1))) v4i*)(void*)g,
        (__attribute__((address_space(3))) v4i*)(void*)l,
        0, 0);
}
__device__ __forceinline__ void wait_async() {
#if __has_builtin(__builtin_amdgcn_s_wait_asynccnt)
    __builtin_amdgcn_s_wait_asynccnt(0);
#else
    asm volatile("s_wait_asynccnt 0x0" ::: "memory");
#endif
}
#endif

// ---------------------------------------------------------------------------
// fp32 -> f16 conversion, 8 elements / thread
// ---------------------------------------------------------------------------
__global__ __launch_bounds__(256)
void cvt_f32_f16(const float* __restrict__ src, f16* __restrict__ dst, int n8) {
    int i = blockIdx.x * blockDim.x + threadIdx.x;
    if (i < n8) {
        const float4 x0 = ((const float4*)src)[i * 2];
        const float4 x1 = ((const float4*)src)[i * 2 + 1];
        v8h o;
        o[0] = (f16)x0.x; o[1] = (f16)x0.y; o[2] = (f16)x0.z; o[3] = (f16)x0.w;
        o[4] = (f16)x1.x; o[5] = (f16)x1.y; o[6] = (f16)x1.z; o[7] = (f16)x1.w;
        ((v8h*)dst)[i] = o;
    }
}

// ---------------------------------------------------------------------------
// C[M,N] = A[M,K] * B[N,K]^T   (A, B row-major f16; C fp32 or f16)
// Block: 256 threads = 8 waves (wave32), waves tiled 2 (M) x 4 (N).
// Block tile 128x256; each wave 64x64 = 4x4 WMMA tiles; K stepped by 32.
// A/B tiles double-buffered in LDS; staged via async global->LDS if available.
// ---------------------------------------------------------------------------
template <bool OUT_F16>
__global__ __launch_bounds__(256)
void gemm_xt(const f16* __restrict__ A, const f16* __restrict__ B,
             void* __restrict__ Cout, int M, int N, int K) {
    __shared__ f16 sA[2][128 * LDSROW];     // 2 x 10 KB
    __shared__ f16 sB[2][256 * LDSROW];     // 2 x 20 KB

    const int tid  = threadIdx.x;
    const int lane = tid & 31;
    const int wave = tid >> 5;
    const int wm   = wave & 1;              // 0..1 along M (64 rows each)
    const int wn   = wave >> 1;             // 0..3 along N (64 cols each)
    const int r    = lane & 15;
    const int hi   = lane >> 4;

    const size_t blockM = (size_t)blockIdx.x * 128;
    const size_t blockN = (size_t)blockIdx.y * 256;

    // Staging: A = 128x32 halves (512 8-half chunks, 2/thread);
    //          B = 256x32 halves (thread t moves all of row t).
    const int ca  = tid * 2;
    const int ar0 = ca >> 2,       ac0 = (ca & 3) * 8;
    const int ar1 = (ca + 1) >> 2, ac1 = ((ca + 1) & 3) * 8;

    const f16* Ab = A + blockM * K;
    const f16* Bb = B + blockN * K;

    v8f acc[4][4] = {};
    int buf = 0;

    // ---- preload K-step 0 into buffer 0 ----
#if ASYNC_LDS
    {
        async_cp16(Ab + (size_t)ar0 * K + ac0, &sA[0][ar0 * LDSROW + ac0]);
        async_cp16(Ab + (size_t)ar1 * K + ac1, &sA[0][ar1 * LDSROW + ac1]);
#pragma unroll
        for (int i = 0; i < 4; ++i)
            async_cp16(Bb + (size_t)tid * K + i * 8, &sB[0][tid * LDSROW + i * 8]);
        wait_async();
    }
#else
    {
        v8h a0 = *(const v8h*)(Ab + (size_t)ar0 * K + ac0);
        v8h a1 = *(const v8h*)(Ab + (size_t)ar1 * K + ac1);
        v8h b0 = *(const v8h*)(Bb + (size_t)tid * K + 0);
        v8h b1 = *(const v8h*)(Bb + (size_t)tid * K + 8);
        v8h b2 = *(const v8h*)(Bb + (size_t)tid * K + 16);
        v8h b3 = *(const v8h*)(Bb + (size_t)tid * K + 24);
        *(v8h*)&sA[0][ar0 * LDSROW + ac0] = a0;
        *(v8h*)&sA[0][ar1 * LDSROW + ac1] = a1;
        *(v8h*)&sB[0][tid * LDSROW + 0]  = b0;
        *(v8h*)&sB[0][tid * LDSROW + 8]  = b1;
        *(v8h*)&sB[0][tid * LDSROW + 16] = b2;
        *(v8h*)&sB[0][tid * LDSROW + 24] = b3;
    }
#endif
    __syncthreads();

    for (int kk = 0; kk < K; kk += BK) {
        const bool more = (kk + BK) < K;

        // ---- prefetch next K-step (overlaps WMMA below) ----
#if ASYNC_LDS
        if (more) {
            const f16* An = Ab + kk + BK;
            const f16* Bn = Bb + kk + BK;
            f16* dA = &sA[buf ^ 1][0];
            f16* dB = &sB[buf ^ 1][0];
            async_cp16(An + (size_t)ar0 * K + ac0, dA + ar0 * LDSROW + ac0);
            async_cp16(An + (size_t)ar1 * K + ac1, dA + ar1 * LDSROW + ac1);
#pragma unroll
            for (int i = 0; i < 4; ++i)
                async_cp16(Bn + (size_t)tid * K + i * 8, dB + tid * LDSROW + i * 8);
        }
#else
        v8h pa0 = {}, pa1 = {}, pb0 = {}, pb1 = {}, pb2 = {}, pb3 = {};
        if (more) {
            const f16* An = Ab + kk + BK;
            const f16* Bn = Bb + kk + BK;
            pa0 = *(const v8h*)(An + (size_t)ar0 * K + ac0);
            pa1 = *(const v8h*)(An + (size_t)ar1 * K + ac1);
            pb0 = *(const v8h*)(Bn + (size_t)tid * K + 0);
            pb1 = *(const v8h*)(Bn + (size_t)tid * K + 8);
            pb2 = *(const v8h*)(Bn + (size_t)tid * K + 16);
            pb3 = *(const v8h*)(Bn + (size_t)tid * K + 24);
        }
#endif

        // ---- fragments from LDS (ISA 7.12.2 layouts) ----
        // A 16x32: lanes 0-15 hold K 0..7 / 16..23, lanes 16-31 K 8..15 / 24..31
        v16h af[4];
#pragma unroll
        for (int i = 0; i < 4; ++i) {
            const f16* p = &sA[buf][(wm * 64 + i * 16 + r) * LDSROW + hi * 8];
            v8h lo = *(const v8h*)p;
            v8h hh = *(const v8h*)(p + 16);
#pragma unroll
            for (int t = 0; t < 8; ++t) { af[i][t] = lo[t]; af[i][8 + t] = hh[t]; }
        }
        // B 32x16: lane n%16 = column n (= row of W); lanes 0-15 K 0..7 / 16..23, lanes 16-31 K 8..15 / 24..31
        v16h bf[4];
#pragma unroll
        for (int j = 0; j < 4; ++j) {
            const f16* p = &sB[buf][(wn * 64 + j * 16 + r) * LDSROW + hi * 8];
            v8h lo = *(const v8h*)p;
            v8h hh = *(const v8h*)(p + 16);
#pragma unroll
            for (int t = 0; t < 8; ++t) { bf[j][t] = lo[t]; bf[j][8 + t] = hh[t]; }
        }

#pragma unroll
        for (int i = 0; i < 4; ++i)
#pragma unroll
            for (int j = 0; j < 4; ++j)
                acc[i][j] = __builtin_amdgcn_wmma_f32_16x16x32_f16(
                    false, af[i], false, bf[j], (short)0, acc[i][j], false, false);

        // ---- make next buffer visible ----
#if ASYNC_LDS
        wait_async();
#else
        if (more) {
            f16* dA = &sA[buf ^ 1][0];
            f16* dB = &sB[buf ^ 1][0];
            *(v8h*)(dA + ar0 * LDSROW + ac0) = pa0;
            *(v8h*)(dA + ar1 * LDSROW + ac1) = pa1;
            *(v8h*)(dB + tid * LDSROW + 0)  = pb0;
            *(v8h*)(dB + tid * LDSROW + 8)  = pb1;
            *(v8h*)(dB + tid * LDSROW + 16) = pb2;
            *(v8h*)(dB + tid * LDSROW + 24) = pb3;
        }
#endif
        __syncthreads();
        buf ^= 1;
    }

    // ---- epilogue: C/D layout VGPR v -> row = hi*8 + v, col = lane%16 ----
#pragma unroll
    for (int i = 0; i < 4; ++i) {
#pragma unroll
        for (int j = 0; j < 4; ++j) {
            const size_t col = blockN + wn * 64 + j * 16 + r;
#pragma unroll
            for (int v = 0; v < 8; ++v) {
                const size_t row = blockM + wm * 64 + i * 16 + hi * 8 + v;
                if (OUT_F16)
                    ((f16*)Cout)[row * N + col] = (f16)acc[i][j][v];
                else
                    ((float*)Cout)[row * N + col] = acc[i][j][v];
            }
        }
    }
}

// ---------------------------------------------------------------------------
// Analytic Gaussian attention as a windowed conv along the sequence.
// All 23 tap weights are compile-time literals (sigma=1): no transcendentals.
// Each thread produces 8 contiguous e (never crosses a head boundary).
// heads: 0,5 center | 1,6 left (c=q-1) | 2,7 right (c=q+1) | 3 first | 4 last
// ---------------------------------------------------------------------------
__global__ __launch_bounds__(256)
void gauss_attn(const f16* __restrict__ V, f16* __restrict__ Att) {
    constexpr float GW[2 * RADIUS + 1] = {
        5.31109225e-27f, 1.92874985e-22f, 2.57675711e-18f, 1.26641655e-14f,
        2.28973485e-11f, 1.52299797e-08f, 3.72665317e-06f, 3.35462628e-04f,
        1.11089965e-02f, 1.35335283e-01f, 6.06530660e-01f, 1.00000000e+00f,
        6.06530660e-01f, 1.35335283e-01f, 1.11089965e-02f, 3.35462628e-04f,
        3.72665317e-06f, 1.52299797e-08f, 2.28973485e-11f, 2.57675711e-18f,
        1.92874985e-22f, 5.31109225e-27f, 5.31109225e-27f };

    const int t  = blockIdx.x * blockDim.x + threadIdx.x;   // one per 8 elems
    const int e0 = (t & 127) * 8;                           // EMBED/8 groups
    const int q  = (t >> 7) & (SEQLEN - 1);
    const int b  = t >> 18;
    const int h  = e0 >> 7;

    int c;
    switch (h) {
        case 0: case 5: c = q;          break;   // center
        case 1: case 6: c = q - 1;      break;   // left  (disp=1)
        case 2: case 7: c = q + 1;      break;   // right (disp=1)
        case 3:         c = 0;          break;   // first
        default:        c = SEQLEN - 1; break;   // last
    }

    const f16* base = V + ((size_t)b * SEQLEN) * EMBED + e0;
    float num[8] = {};
    float den = 0.0f;
#pragma unroll
    for (int d = -RADIUS; d <= RADIUS; ++d) {
        const int k = c + d;
        if (k >= 0 && k < SEQLEN) {
            const float w = GW[d + RADIUS];      // literal after unroll
            den += w;
            const v8h vv = *(const v8h*)(base + (size_t)k * EMBED);
#pragma unroll
            for (int u = 0; u < 8; ++u) num[u] += w * (float)vv[u];
        }
    }
    const float inv = 1.0f / den;               // den > 0 always
    v8h o;
#pragma unroll
    for (int u = 0; u < 8; ++u) o[u] = (f16)(num[u] * inv);
    ((v8h*)Att)[t] = o;
}

// ---------------------------------------------------------------------------
extern "C" void kernel_launch(void* const* d_in, const int* in_sizes, int n_in,
                              void* d_out, int out_size, void* d_ws, size_t ws_size,
                              hipStream_t stream) {
    const float* x     = (const float*)d_in[0];
    const float* W_in  = (const float*)d_in[1];
    const float* W_out = (const float*)d_in[2];
    float* out = (float*)d_out;

    const int M  = BATCH * SEQLEN;           // 16384
    const int E  = EMBED;                    // 1024
    const int nX = M * E;                    // 16,777,216
    const int nW = E * E;                    // 1,048,576

    // workspace layout (f16): x | W_in | W_out | V | attended  (~100 MB)
    f16* xh  = (f16*)d_ws;
    f16* wih = xh  + (size_t)nX;
    f16* woh = wih + (size_t)nW;
    f16* vh  = woh + (size_t)nW;
    f16* ah  = vh  + (size_t)nX;

    const int T = 256;
    cvt_f32_f16<<<(nX / 8) / T, T, 0, stream>>>(x,     xh,  nX / 8);
    cvt_f32_f16<<<(nW / 8) / T, T, 0, stream>>>(W_in,  wih, nW / 8);
    cvt_f32_f16<<<(nW / 8) / T, T, 0, stream>>>(W_out, woh, nW / 8);

    dim3 grid(M / 128, E / 256);             // (128, 4)
    gemm_xt<true ><<<grid, 256, 0, stream>>>(xh, wih, vh, M, E, E);

    gauss_attn<<<(nX / 8) / T, T, 0, stream>>>(vh, ah);

    gemm_xt<false><<<grid, 256, 0, stream>>>(ah, woh, out, M, E, E);
}
